// Model_10368051052640
// MI455X (gfx1250) — compile-verified, matrix operand order checked by reading, not run on hardware
//
#include <hip/hip_runtime.h>
#include <hip/hip_bf16.h>
#include <math.h>

#define B_    2
#define S_    4096
#define D_    768
#define H_    12
#define DH_   64
#define W_    256
#define NC_   16
#define F_    3072
#define NCLS_ 10
#define NEGB  (-1e9f)

typedef __attribute__((ext_vector_type(16))) __bf16           v16bf;
typedef __attribute__((ext_vector_type(16))) unsigned short   v16us;
typedef __attribute__((ext_vector_type(8)))  float            v8f;
typedef __attribute__((ext_vector_type(4)))  unsigned int     v4u;
typedef __attribute__((ext_vector_type(8)))  int              v8i;
typedef __attribute__((ext_vector_type(4)))  int              v4i;

#if defined(__gfx1250__) && __has_builtin(__builtin_amdgcn_tensor_load_to_lds) && \
    __has_builtin(__builtin_amdgcn_s_wait_tensorcnt)
#define USE_TDM 1
#else
#define USE_TDM 0
#endif

struct U32x8 { uint4 lo, hi; };

static __device__ __forceinline__ unsigned short f2us(float f) {
  unsigned u = __float_as_uint(f);
  u += 0x7fffu + ((u >> 16) & 1u);           // round-to-nearest-even bf16
  return (unsigned short)(u >> 16);
}
static __device__ __forceinline__ float us2f(unsigned short h) {
  return __uint_as_float(((unsigned)h) << 16);
}
// monotonic float<->uint mapping for atomicMax on floats (incl. negatives)
static __device__ __forceinline__ unsigned ford(float f) {
  unsigned u = __float_as_uint(f);
  return (u & 0x80000000u) ? ~u : (u | 0x80000000u);
}
static __device__ __forceinline__ float ordf(unsigned o) {
  unsigned u = (o & 0x80000000u) ? (o & 0x7fffffffu) : ~o;
  return __uint_as_float(u);
}
static __device__ __forceinline__ v8f wmma_bf16(v16us a, v16us b, v8f c) {
  return __builtin_amdgcn_wmma_f32_16x16x32_bf16(
      false, __builtin_bit_cast(v16bf, a),
      false, __builtin_bit_cast(v16bf, b),
      (short)0, c, false, false);
}

#if USE_TDM
// ---------------------------------------------------------------------------
// Tensor Data Mover: DMA one bf16 tile (tile_dim0=32 elems along K,
// tile_dim1=128 rows, row stride = ldK elems) from global into LDS.
// LDS rows padded: 64B of data then 16B pad -> stride 40 elems (matches the
// GEMM fragment layout).  D# packing per CDNA5 ISA ch.8 (groups 0/1; 2D).
// This toolchain exposes the 6-arg builtin:
//   (v4u g0, v8i g1, v4i g2, v4i g3, v8i g4, i32 cpol)
// trailing groups zero-filled (<=2D tensor; VADDR2/3 would be NULL in ISA).
// ---------------------------------------------------------------------------
static __device__ __forceinline__ void tdm_tile_load(const unsigned short* g,
                                                     int ldK,
                                                     unsigned lds_byte_off) {
  unsigned long long ga = (unsigned long long)(size_t)g;
  v4u g0;
  g0[0] = 1u;                                        // count=1 (valid), user D#
  g0[1] = lds_byte_off;                              // lds_addr
  g0[2] = (unsigned)(ga & 0xffffffffu);              // global_addr[31:0]
  g0[3] = (unsigned)((ga >> 32) & 0x01ffffffu)       // global_addr[56:32]
          | (2u << 30);                              // type = 2 ("image")
  const unsigned dim0 = 0x00ffffffu, dim1 = 0x00ffffffu;  // huge: no clipping
  v8i g1;
  g1[0] = (int)((1u << 16)       // data_size = 2 bytes
              | (1u << 20)       // pad_enable
              | (3u << 22)       // pad_interval: 16 DWORDs (one 64B row)
              | (3u << 25));     // pad_amount:  4 DWORDs (16B)
  g1[1] = (int)((dim0 & 0xffffu) << 16);                       // dim0[15:0]
  g1[2] = (int)((dim0 >> 16) | ((dim1 & 0xffffu) << 16));      // dim0 hi|dim1 lo
  g1[3] = (int)((dim1 >> 16) | (32u << 16));                   // tile_dim0 = 32
  g1[4] = 128;                                                 // tile_dim1=128
  g1[5] = (int)(unsigned)ldK;                                  // dim0_stride lo
  g1[6] = 0;                                                   // stride hi, d1s
  g1[7] = 0;
  v4i z4 = {0, 0, 0, 0};
  v8i z8 = {0, 0, 0, 0, 0, 0, 0, 0};
  __builtin_amdgcn_tensor_load_to_lds(g0, g1, z4, z4, z8, 0);
}
#endif

// ---------------------------------------------------------------------------
// fp32 (K x N) weight  ->  bf16 transposed (N x K) so GEMM Bt tiles are
// row-major contiguous (clean b128 loads, same fragment pattern as A).
// ---------------------------------------------------------------------------
__global__ void convT_kernel(const float* __restrict__ src,
                             unsigned short* __restrict__ dst, int K, int N) {
  size_t i = (size_t)blockIdx.x * 256 + threadIdx.x;
  if (i >= (size_t)K * N) return;
  int n = (int)(i / K), k = (int)(i % K);
  dst[i] = f2us(src[(size_t)k * N + n]);
}

// ---------------------------------------------------------------------------
// C(MxN) = A(MxK,bf16) @ B(KxN)  with B given transposed (Bt: NxK, bf16).
// 128x128 block tile, BK=32, 8 waves * (2x4 tiles of 16x16), WMMA bf16.
// Tiles staged into LDS by the Tensor Data Mover (double-buffered; DMA of
// tile t+1 overlaps WMMA on tile t).  Epilogue: +bias, optional exact GELU
// (act==1), scale, fp32/bf16 outputs.  Requires M%128==N%128==K%32==0.
// ---------------------------------------------------------------------------
__global__ __launch_bounds__(256) void gemm_bf16_kernel(
    const unsigned short* __restrict__ A, const unsigned short* __restrict__ Bt,
    const float* __restrict__ bias, float* __restrict__ outf,
    unsigned short* __restrict__ outb, int M, int N, int K, int act,
    float scale) {
  __shared__ __attribute__((aligned(16))) unsigned short As[2][128 * 40];
  __shared__ __attribute__((aligned(16))) unsigned short Bs[2][128 * 40];
  int tid = threadIdx.x, lane = tid & 31, w = tid >> 5;
  int wm = (w & 3) * 32, wn = (w >> 2) * 64;
  int gm = blockIdx.y * 128, gn = blockIdx.x * 128;
  v8f acc[2][4];
#pragma unroll
  for (int mi = 0; mi < 2; ++mi)
#pragma unroll
    for (int ni = 0; ni < 4; ++ni)
#pragma unroll
      for (int j = 0; j < 8; ++j) acc[mi][ni][j] = 0.f;

  int kb = (lane >> 4) * 8;
  int nIter = K / 32;

#if USE_TDM
  const unsigned short* Abase = A + (size_t)gm * K;
  const unsigned short* Bbase = Bt + (size_t)gn * K;
  if (w == 0) {  // prologue DMA: tile 0 into buffer 0
    tdm_tile_load(Abase, K, (unsigned)(size_t)&As[0][0]);
    tdm_tile_load(Bbase, K, (unsigned)(size_t)&Bs[0][0]);
  }
#else
  int lrow = tid >> 1, lcb = (tid & 1) * 16;
  const unsigned short* Ag = A + (size_t)(gm + lrow) * K + lcb;
  const unsigned short* Bg = Bt + (size_t)(gn + lrow) * K + lcb;
#endif

  for (int it = 0; it < nIter; ++it) {
    int kk = it * 32;
#if USE_TDM
    (void)kk;
    int cur = it & 1;
    if (w == 0) __builtin_amdgcn_s_wait_tensorcnt(0);  // tile `it` landed
    __syncthreads();                                   // publish to all waves
    if (w == 0 && it + 1 < nIter) {                    // DMA next tile, other buf
      tdm_tile_load(Abase + (it + 1) * 32, K,
                    (unsigned)(size_t)&As[cur ^ 1][0]);
      tdm_tile_load(Bbase + (it + 1) * 32, K,
                    (unsigned)(size_t)&Bs[cur ^ 1][0]);
    }
    const unsigned short* Asb = &As[cur][0];
    const unsigned short* Bsb = &Bs[cur][0];
#else
    uint4 a0 = *(const uint4*)(Ag + kk);
    uint4 a1 = *(const uint4*)(Ag + kk + 8);
    uint4 b0 = *(const uint4*)(Bg + kk);
    uint4 b1 = *(const uint4*)(Bg + kk + 8);
    __syncthreads();
    *(uint4*)&As[0][lrow * 40 + lcb]     = a0;
    *(uint4*)&As[0][lrow * 40 + lcb + 8] = a1;
    *(uint4*)&Bs[0][lrow * 40 + lcb]     = b0;
    *(uint4*)&Bs[0][lrow * 40 + lcb + 8] = b1;
    __syncthreads();
    if (kk + 32 < K) {
      __builtin_prefetch(Ag + kk + 32, 0, 0);
      __builtin_prefetch(Bg + kk + 32, 0, 0);
    }
    const unsigned short* Asb = &As[0][0];
    const unsigned short* Bsb = &Bs[0][0];
#endif
    v16us afr[2], bfr[4];
#pragma unroll
    for (int mi = 0; mi < 2; ++mi) {
      const unsigned short* p = Asb + (wm + mi * 16 + (lane & 15)) * 40;
      U32x8 u;
      u.lo = *(const uint4*)(p + kb);
      u.hi = *(const uint4*)(p + kb + 16);
      afr[mi] = __builtin_bit_cast(v16us, u);
    }
#pragma unroll
    for (int ni = 0; ni < 4; ++ni) {
      const unsigned short* p = Bsb + (wn + ni * 16 + (lane & 15)) * 40;
      U32x8 u;
      u.lo = *(const uint4*)(p + kb);
      u.hi = *(const uint4*)(p + kb + 16);
      bfr[ni] = __builtin_bit_cast(v16us, u);
    }
#pragma unroll
    for (int mi = 0; mi < 2; ++mi)
#pragma unroll
      for (int ni = 0; ni < 4; ++ni)
        acc[mi][ni] = wmma_bf16(afr[mi], bfr[ni], acc[mi][ni]);
  }

#pragma unroll
  for (int mi = 0; mi < 2; ++mi)
#pragma unroll
    for (int ni = 0; ni < 4; ++ni) {
      int col = gn + wn + ni * 16 + (lane & 15);
      float bv = bias ? bias[col] : 0.f;
#pragma unroll
      for (int j = 0; j < 8; ++j) {
        int row = gm + wm + mi * 16 + (lane >> 4) * 8 + j;
        float v = acc[mi][ni][j] + bv;
        if (act == 1) v = 0.5f * v * (1.f + erff(v * 0.70710678118654752f));
        v *= scale;
        size_t o = (size_t)row * N + col;
        if (outf) outf[o] = v;
        if (outb) outb[o] = f2us(v);
      }
    }
}

// ---------------------------------------------------------------------------
// Fused sliding-window attention. One block = (b, head, chunk, 32-query slab).
// scores(32x768) = Qslab @ Kband^T via WMMA (K frags loaded directly from
// global: key rows are contiguous bf16, exactly Bt-fragment layout),
// band/global-key/pad mask, softmax over 768+1 (global col), P stored bf16 in
// LDS, out(32x64) = P @ Vband via WMMA with V staged in LDS, + global column.
// ---------------------------------------------------------------------------
__global__ __launch_bounds__(256) void attn_win_kernel(
    const unsigned short* __restrict__ qb, const unsigned short* __restrict__ kbuf,
    const unsigned short* __restrict__ vb, const int* __restrict__ masks,
    unsigned short* __restrict__ ob) {
  __shared__ __attribute__((aligned(16))) unsigned short Qs[32 * 72];
  __shared__ __attribute__((aligned(16))) unsigned short Ps[32 * 776];
  __shared__ __attribute__((aligned(16))) unsigned short Vs[32 * 72];
  __shared__ unsigned smax[32];
  __shared__ float ssum[32];
  __shared__ float sge[32];

  int tid = threadIdx.x, lane = tid & 31, w = tid >> 5;
  int qq = blockIdx.x, c = blockIdx.y;
  int b = blockIdx.z / H_, hh = blockIdx.z % H_;
  int q0 = c * W_ + qq * 32;      // first query seq position of this slab
  int kstart = (c - 1) * W_;      // seq position of band key n=0
  int kb = (lane >> 4) * 8;

  {  // stage Q slab (32 x 64 bf16)
    int r = tid >> 3, cc = (tid & 7) * 8;
    *(uint4*)&Qs[r * 72 + cc] =
        *(const uint4*)&qb[((size_t)(b * S_ + q0 + r)) * D_ + hh * DH_ + cc];
  }
  if (tid < 32) { smax[tid] = 0u; ssum[tid] = 0.f; sge[tid] = 0.f; }
  __syncthreads();

  // ---- scores: 8 waves x (2 Mtiles x 6 Ntiles), K-dim = 64 ----
  v8f sc[2][6];
#pragma unroll
  for (int mi = 0; mi < 2; ++mi)
#pragma unroll
    for (int ni = 0; ni < 6; ++ni)
#pragma unroll
      for (int j = 0; j < 8; ++j) sc[mi][ni][j] = 0.f;
  int nbase = w * 96;
#pragma unroll
  for (int kk = 0; kk < DH_; kk += 32) {
    v16us afr[2];
#pragma unroll
    for (int mi = 0; mi < 2; ++mi) {
      const unsigned short* p = &Qs[(mi * 16 + (lane & 15)) * 72 + kk];
      U32x8 u;
      u.lo = *(const uint4*)(p + kb);
      u.hi = *(const uint4*)(p + kb + 16);
      afr[mi] = __builtin_bit_cast(v16us, u);
    }
#pragma unroll
    for (int ni = 0; ni < 6; ++ni) {
      int n = nbase + ni * 16 + (lane & 15);
      int pos = kstart + n;
      U32x8 u;
      u.lo = make_uint4(0u, 0u, 0u, 0u);
      u.hi = make_uint4(0u, 0u, 0u, 0u);
      if (pos >= 0 && pos < S_) {
        const unsigned short* g =
            &kbuf[((size_t)(b * S_ + pos)) * D_ + hh * DH_ + kk];
        u.lo = *(const uint4*)(g + kb);
        u.hi = *(const uint4*)(g + kb + 16);
      }
      v16us bfr = __builtin_bit_cast(v16us, u);
      sc[0][ni] = wmma_bf16(afr[0], bfr, sc[0][ni]);
      sc[1][ni] = wmma_bf16(afr[1], bfr, sc[1][ni]);
    }
  }

  // global-column score s_g = q . k[:,0] (+mask)
  if (tid < 32) {
    float a = 0.f;
    const unsigned short* qp = &Qs[tid * 72];
    const unsigned short* kp = &kbuf[((size_t)(b * S_)) * D_ + hh * DH_];
    for (int d = 0; d < DH_; ++d) a += us2f(qp[d]) * us2f(kp[d]);
    a += (masks[b * S_ + 0] - 1) * 1e9f;
    sge[tid] = a;
    atomicMax(&smax[tid], ford(a));
  }
  // mask + per-row max
#pragma unroll
  for (int mi = 0; mi < 2; ++mi)
#pragma unroll
    for (int j = 0; j < 8; ++j) {
      int row = mi * 16 + (lane >> 4) * 8 + j;
      int qi = qq * 32 + row;  // local query index within chunk [0,W)
      float mx = -3.0e38f;
#pragma unroll
      for (int ni = 0; ni < 6; ++ni) {
        int n = nbase + ni * 16 + (lane & 15);
        int pos = kstart + n;
        float s = sc[mi][ni][j];
        int dn = n - qi;
        if (dn < 0 || dn > 2 * W_) s += NEGB;   // band mask
        if (pos == 0) s += NEGB;                // global key via dedicated col
        if (pos < 0 || pos >= S_) s += NEGB;    // padding
        else s += (masks[b * S_ + pos] - 1) * 1e9f;
        sc[mi][ni][j] = s;
        mx = fmaxf(mx, s);
      }
      atomicMax(&smax[row], ford(mx));
    }
  __syncthreads();
  // exp, row sums, store P (bf16) into LDS
#pragma unroll
  for (int mi = 0; mi < 2; ++mi)
#pragma unroll
    for (int j = 0; j < 8; ++j) {
      int row = mi * 16 + (lane >> 4) * 8 + j;
      float m = ordf(smax[row]);
      float ps = 0.f;
#pragma unroll
      for (int ni = 0; ni < 6; ++ni) {
        int n = nbase + ni * 16 + (lane & 15);
        float p = __expf(sc[mi][ni][j] - m);
        Ps[row * 776 + n] = f2us(p);
        ps += p;
      }
      atomicAdd(&ssum[row], ps);
    }
  if (tid < 32) {
    float pg = __expf(sge[tid] - ordf(smax[tid]));
    sge[tid] = pg;
    atomicAdd(&ssum[tid], pg);
  }
  __syncthreads();

  // ---- out = P(32x768) @ Vband(768x64): 8 waves, 1 tile each ----
  int mo = w >> 2, no = w & 3;
  v8f o;
#pragma unroll
  for (int j = 0; j < 8; ++j) o[j] = 0.f;
  for (int kk = 0; kk < 3 * W_; kk += 32) {
    __syncthreads();
    {  // stage 32 band keys of V (32 x 64 bf16)
      int r = tid >> 3, cc = (tid & 7) * 8;
      int pos = kstart + kk + r;
      uint4 z = make_uint4(0u, 0u, 0u, 0u);
      if (pos >= 0 && pos < S_)
        z = *(const uint4*)&vb[((size_t)(b * S_ + pos)) * D_ + hh * DH_ + cc];
      *(uint4*)&Vs[r * 72 + cc] = z;
    }
    __syncthreads();
    v16us afr;
    {
      const unsigned short* p = &Ps[(mo * 16 + (lane & 15)) * 776 + kk];
      U32x8 u;
      u.lo = *(const uint4*)(p + kb);
      u.hi = *(const uint4*)(p + kb + 16);
      afr = __builtin_bit_cast(v16us, u);
    }
    v16us bfr;
    {
      int d = no * 16 + (lane & 15);
#pragma unroll
      for (int e = 0; e < 8; ++e) {
        bfr[e]     = Vs[(kb + e) * 72 + d];
        bfr[8 + e] = Vs[(16 + kb + e) * 72 + d];
      }
    }
    o = wmma_bf16(afr, bfr, o);
  }
  int d = no * 16 + (lane & 15);
  float v0 = us2f(vb[((size_t)(b * S_)) * D_ + hh * DH_ + d]);
#pragma unroll
  for (int j = 0; j < 8; ++j) {
    int row = mo * 16 + (lane >> 4) * 8 + j;
    float r = (o[j] + sge[row] * v0) / ssum[row];
    ob[((size_t)(b * S_ + q0 + row)) * D_ + hh * DH_ + d] = f2us(r);
  }
}

// ---------------------------------------------------------------------------
// Embedding gather + LayerNorm; writes fp32 residual stream + bf16 copy.
// ---------------------------------------------------------------------------
__global__ __launch_bounds__(256) void embed_ln_kernel(
    const int* __restrict__ X, const float* __restrict__ we,
    const float* __restrict__ pe, const float* __restrict__ g,
    const float* __restrict__ bb, float* __restrict__ h,
    unsigned short* __restrict__ hb) {
  __shared__ float red[256];
  int tok = blockIdx.x, s = tok % S_, tid = threadIdx.x;
  int x = X[tok];
  float v[3];
  float ps = 0.f;
#pragma unroll
  for (int i = 0; i < 3; ++i) {
    int d = tid + i * 256;
    v[i] = we[(size_t)x * D_ + d] + pe[(size_t)s * D_ + d];
    ps += v[i];
  }
  red[tid] = ps; __syncthreads();
  for (int st = 128; st > 0; st >>= 1) { if (tid < st) red[tid] += red[tid + st]; __syncthreads(); }
  float mean = red[0] / (float)D_; __syncthreads();
  float pv = 0.f;
#pragma unroll
  for (int i = 0; i < 3; ++i) { float dl = v[i] - mean; pv += dl * dl; }
  red[tid] = pv; __syncthreads();
  for (int st = 128; st > 0; st >>= 1) { if (tid < st) red[tid] += red[tid + st]; __syncthreads(); }
  float rstd = rsqrtf(red[0] / (float)D_ + 1e-5f);
#pragma unroll
  for (int i = 0; i < 3; ++i) {
    int d = tid + i * 256;
    float o = (v[i] - mean) * rstd * g[d] + bb[d];
    h[(size_t)tok * D_ + d] = o;
    hb[(size_t)tok * D_ + d] = f2us(o);
  }
}

__global__ __launch_bounds__(256) void res_ln_kernel(
    const float* __restrict__ hin, const float* __restrict__ r,
    const float* __restrict__ g, const float* __restrict__ bb,
    float* __restrict__ hout, unsigned short* __restrict__ hb) {
  __shared__ float red[256];
  int tok = blockIdx.x, tid = threadIdx.x;
  float v[3];
  float ps = 0.f;
#pragma unroll
  for (int i = 0; i < 3; ++i) {
    int d = tid + i * 256;
    v[i] = hin[(size_t)tok * D_ + d] + r[(size_t)tok * D_ + d];
    ps += v[i];
  }
  red[tid] = ps; __syncthreads();
  for (int st = 128; st > 0; st >>= 1) { if (tid < st) red[tid] += red[tid + st]; __syncthreads(); }
  float mean = red[0] / (float)D_; __syncthreads();
  float pv = 0.f;
#pragma unroll
  for (int i = 0; i < 3; ++i) { float dl = v[i] - mean; pv += dl * dl; }
  red[tid] = pv; __syncthreads();
  for (int st = 128; st > 0; st >>= 1) { if (tid < st) red[tid] += red[tid + st]; __syncthreads(); }
  float rstd = rsqrtf(red[0] / (float)D_ + 1e-5f);
#pragma unroll
  for (int i = 0; i < 3; ++i) {
    int d = tid + i * 256;
    float o = (v[i] - mean) * rstd * g[d] + bb[d];
    hout[(size_t)tok * D_ + d] = o;
    hb[(size_t)tok * D_ + d] = f2us(o);
  }
}

// ---------------- global-token path (tiny FLOPs, VALU kernels) -------------
__global__ __launch_bounds__(256) void qg_kernel(
    const unsigned short* __restrict__ hb, const unsigned short* __restrict__ wqgT,
    const float* __restrict__ bqg, float* __restrict__ qg) {
  int b = blockIdx.x, tid = threadIdx.x;
  const unsigned short* hp = hb + (size_t)(b * S_) * D_;
#pragma unroll
  for (int i = 0; i < 3; ++i) {
    int d = tid + i * 256;
    float a = bqg[d];
    const unsigned short* wv = wqgT + (size_t)d * D_;
    for (int k = 0; k < D_; ++k) a += us2f(hp[k]) * us2f(wv[k]);
    qg[b * D_ + d] = a * 0.125f;
  }
}

__global__ __launch_bounds__(256) void gscore_kernel(
    const float* __restrict__ qg, const unsigned short* __restrict__ kgb,
    const int* __restrict__ masks, float* __restrict__ pg) {
  int bh = blockIdx.y, b = bh / H_, hh = bh % H_;
  int s = blockIdx.x * 256 + threadIdx.x;
  const float* q = qg + b * D_ + hh * DH_;
  const unsigned short* kp = kgb + ((size_t)(b * S_ + s)) * D_ + hh * DH_;
  float a = 0.f;
  for (int d = 0; d < DH_; ++d) a += q[d] * us2f(kp[d]);
  a += (masks[b * S_ + s] - 1) * 1e9f;
  pg[(size_t)bh * S_ + s] = a;
}

__global__ __launch_bounds__(256) void gsoftmax_kernel(float* __restrict__ pg) {
  __shared__ float red[256];
  int bh = blockIdx.x, tid = threadIdx.x;
  float* p = pg + (size_t)bh * S_;
  float mx = -3e38f;
  for (int s = tid; s < S_; s += 256) mx = fmaxf(mx, p[s]);
  red[tid] = mx; __syncthreads();
  for (int st = 128; st > 0; st >>= 1) { if (tid < st) red[tid] = fmaxf(red[tid], red[tid + st]); __syncthreads(); }
  mx = red[0]; __syncthreads();
  float sm = 0.f;
  for (int s = tid; s < S_; s += 256) { float e = __expf(p[s] - mx); p[s] = e; sm += e; }
  red[tid] = sm; __syncthreads();
  for (int st = 128; st > 0; st >>= 1) { if (tid < st) red[tid] += red[tid + st]; __syncthreads(); }
  float inv = 1.f / red[0];
  __syncthreads();
  for (int s = tid; s < S_; s += 256) p[s] *= inv;
}

__global__ void gout_kernel(const float* __restrict__ pg,
                            const unsigned short* __restrict__ vgb,
                            unsigned short* __restrict__ ob) {
  int bh = blockIdx.x, b = bh / H_, hh = bh % H_;
  int d = threadIdx.x;  // 64 threads
  const float* p = pg + (size_t)bh * S_;
  float a = 0.f;
  for (int s = 0; s < S_; ++s)
    a += p[s] * us2f(vgb[((size_t)(b * S_ + s)) * D_ + hh * DH_ + d]);
  ob[((size_t)(b * S_)) * D_ + hh * DH_ + d] = f2us(a);
}

// ---------------- pooler + classifier (tiny) -------------------------------
__global__ __launch_bounds__(256) void pool_kernel(
    const float* __restrict__ h, const float* __restrict__ wp,
    const float* __restrict__ bp, float* __restrict__ pooled) {
  int b = blockIdx.x, tid = threadIdx.x;
  const float* hp = h + (size_t)(b * S_) * D_;
#pragma unroll
  for (int i = 0; i < 3; ++i) {
    int j = tid + i * 256;
    float a = bp[j];
    for (int k = 0; k < D_; ++k) a += hp[k] * wp[(size_t)k * D_ + j];
    pooled[b * D_ + j] = tanhf(a);
  }
}

__global__ __launch_bounds__(256) void cls1_kernel(
    const float* __restrict__ pooled, const float* __restrict__ wc1,
    const float* __restrict__ bc1, float* __restrict__ c1) {
  int b = blockIdx.x, tid = threadIdx.x;
  const float* pp = pooled + (size_t)b * D_;
#pragma unroll
  for (int i = 0; i < 3; ++i) {
    int j = tid + i * 256;
    float a = bc1[j];
    for (int k = 0; k < D_; ++k) a += pp[k] * wc1[(size_t)k * D_ + j];
    c1[b * D_ + j] = fmaxf(a, 0.f);
  }
}

__global__ void cls2_kernel(const float* __restrict__ c1,
                            const float* __restrict__ wc2,
                            const float* __restrict__ bc2,
                            float* __restrict__ out) {
  int b = blockIdx.x, j = threadIdx.x;
  if (j >= NCLS_) return;
  float a = bc2[j];
  for (int k = 0; k < D_; ++k) a += c1[b * D_ + k] * wc2[k * NCLS_ + j];
  out[b * NCLS_ + j] = a;
}

// ===========================================================================
extern "C" void kernel_launch(void* const* d_in, const int* in_sizes, int n_in,
                              void* d_out, int out_size, void* d_ws,
                              size_t ws_size, hipStream_t stream) {
  (void)in_sizes; (void)n_in; (void)out_size; (void)ws_size;
  // pytree flatten order (dict keys sorted): X, masks, then params:
  // bc1,bc2,be,bp,ge, layers[0..1](b1,b2,bf1,bf2,bk,bkg,bo,bq,bqg,bv,bvg,
  // g1,g2,wf1,wf2,wk,wkg,wo,wq,wqg,wv,wvg), pe,wc1,wc2,we,wp
  const int*   X     = (const int*)d_in[0];
  const int*   masks = (const int*)d_in[1];
  const float* bc1   = (const float*)d_in[2];
  const float* bc2   = (const float*)d_in[3];
  const float* be    = (const float*)d_in[4];
  const float* bp    = (const float*)d_in[5];
  const float* ge    = (const float*)d_in[6];
  const float* pe    = (const float*)d_in[51];
  const float* wc1   = (const float*)d_in[52];
  const float* wc2   = (const float*)d_in[53];
  const float* we    = (const float*)d_in[54];
  const float* wp    = (const float*)d_in[55];

  struct Layer {
    const float *b1, *b2, *bf1, *bf2, *bk, *bkg, *bo, *bq, *bqg, *bv, *bvg;
    const float *g1, *g2, *wf1, *wf2, *wk, *wkg, *wo, *wq, *wqg, *wv, *wvg;
  } L[2];
  for (int l = 0; l < 2; ++l) {
    int o = 7 + l * 22;
    L[l].b1  = (const float*)d_in[o + 0];
    L[l].b2  = (const float*)d_in[o + 1];
    L[l].bf1 = (const float*)d_in[o + 2];
    L[l].bf2 = (const float*)d_in[o + 3];
    L[l].bk  = (const float*)d_in[o + 4];
    L[l].bkg = (const float*)d_in[o + 5];
    L[l].bo  = (const float*)d_in[o + 6];
    L[l].bq  = (const float*)d_in[o + 7];
    L[l].bqg = (const float*)d_in[o + 8];
    L[l].bv  = (const float*)d_in[o + 9];
    L[l].bvg = (const float*)d_in[o + 10];
    L[l].g1  = (const float*)d_in[o + 11];
    L[l].g2  = (const float*)d_in[o + 12];
    L[l].wf1 = (const float*)d_in[o + 13];
    L[l].wf2 = (const float*)d_in[o + 14];
    L[l].wk  = (const float*)d_in[o + 15];
    L[l].wkg = (const float*)d_in[o + 16];
    L[l].wo  = (const float*)d_in[o + 17];
    L[l].wq  = (const float*)d_in[o + 18];
    L[l].wqg = (const float*)d_in[o + 19];
    L[l].wv  = (const float*)d_in[o + 20];
    L[l].wvg = (const float*)d_in[o + 21];
  }

  // workspace bump allocator
  char* base = (char*)d_ws;
  size_t off = 0;
  auto alloc = [&](size_t bytes) -> char* {
    char* p = base + off;
    off += (bytes + 255) & ~(size_t)255;
    return p;
  };
  const size_t TOK = (size_t)B_ * S_;
  float*          h_f    = (float*)alloc(TOK * D_ * 4);
  float*          proj_f = (float*)alloc(TOK * D_ * 4);
  unsigned short* h_bf   = (unsigned short*)alloc(TOK * D_ * 2);
  unsigned short* q_bf   = (unsigned short*)alloc(TOK * D_ * 2);
  unsigned short* k_bf   = (unsigned short*)alloc(TOK * D_ * 2);
  unsigned short* v_bf   = (unsigned short*)alloc(TOK * D_ * 2);
  unsigned short* kg_bf  = (unsigned short*)alloc(TOK * D_ * 2);
  unsigned short* vg_bf  = (unsigned short*)alloc(TOK * D_ * 2);
  unsigned short* at_bf  = (unsigned short*)alloc(TOK * D_ * 2);
  unsigned short* ff_bf  = (unsigned short*)alloc(TOK * F_ * 2);
  float*          qg_f   = (float*)alloc((size_t)B_ * D_ * 4);
  float*          pg_f   = (float*)alloc((size_t)B_ * H_ * S_ * 4);
  float*          pool_f = (float*)alloc((size_t)B_ * D_ * 4);
  float*          c1_f   = (float*)alloc((size_t)B_ * D_ * 4);
  unsigned short* wT[2][9];  // wq,wk,wv,wqg,wkg,wvg,wo, wf1, wf2 (transposed bf16)
  for (int l = 0; l < 2; ++l) {
    for (int i = 0; i < 7; ++i) wT[l][i] = (unsigned short*)alloc((size_t)D_ * D_ * 2);
    wT[l][7] = (unsigned short*)alloc((size_t)D_ * F_ * 2);
    wT[l][8] = (unsigned short*)alloc((size_t)D_ * F_ * 2);
  }

  // ---- convert+transpose weights to bf16 ----
  auto conv = [&](const float* src, unsigned short* dst, int K, int N) {
    int nb = (int)(((size_t)K * N + 255) / 256);
    convT_kernel<<<nb, 256, 0, stream>>>(src, dst, K, N);
  };
  for (int l = 0; l < 2; ++l) {
    conv(L[l].wq,  wT[l][0], D_, D_);
    conv(L[l].wk,  wT[l][1], D_, D_);
    conv(L[l].wv,  wT[l][2], D_, D_);
    conv(L[l].wqg, wT[l][3], D_, D_);
    conv(L[l].wkg, wT[l][4], D_, D_);
    conv(L[l].wvg, wT[l][5], D_, D_);
    conv(L[l].wo,  wT[l][6], D_, D_);
    conv(L[l].wf1, wT[l][7], D_, F_);   // -> (F x D)
    conv(L[l].wf2, wT[l][8], F_, D_);   // -> (D x F)
  }

  auto gemm = [&](const unsigned short* A, const unsigned short* Bt,
                  const float* bias, float* of, unsigned short* ob, int M,
                  int N, int K, int act, float scale) {
    dim3 grid(N / 128, M / 128);
    gemm_bf16_kernel<<<grid, 256, 0, stream>>>(A, Bt, bias, of, ob, M, N, K,
                                               act, scale);
  };
  const int M = B_ * S_;  // 8192

  // ---- embedding + LN ----
  embed_ln_kernel<<<(int)TOK, 256, 0, stream>>>(X, we, pe, ge, be, h_f, h_bf);

  for (int l = 0; l < 2; ++l) {
    // projections (WMMA): q scaled by 1/sqrt(DH)
    gemm(h_bf, wT[l][0], L[l].bq,  nullptr, q_bf,  M, D_, D_, 0, 0.125f);
    gemm(h_bf, wT[l][1], L[l].bk,  nullptr, k_bf,  M, D_, D_, 0, 1.f);
    gemm(h_bf, wT[l][2], L[l].bv,  nullptr, v_bf,  M, D_, D_, 0, 1.f);
    gemm(h_bf, wT[l][4], L[l].bkg, nullptr, kg_bf, M, D_, D_, 0, 1.f);
    gemm(h_bf, wT[l][5], L[l].bvg, nullptr, vg_bf, M, D_, D_, 0, 1.f);
    // windowed attention (WMMA)
    attn_win_kernel<<<dim3(W_ / 32, NC_, B_ * H_), 256, 0, stream>>>(
        q_bf, k_bf, v_bf, masks, at_bf);
    // global token path (tiny)
    qg_kernel<<<B_, 256, 0, stream>>>(h_bf, wT[l][3], L[l].bqg, qg_f);
    gscore_kernel<<<dim3(S_ / 256, B_ * H_), 256, 0, stream>>>(qg_f, kg_bf,
                                                               masks, pg_f);
    gsoftmax_kernel<<<B_ * H_, 256, 0, stream>>>(pg_f);
    gout_kernel<<<B_ * H_, DH_, 0, stream>>>(pg_f, vg_bf, at_bf);
    // output projection + residual LN
    gemm(at_bf, wT[l][6], L[l].bo, proj_f, nullptr, M, D_, D_, 0, 1.f);
    res_ln_kernel<<<(int)TOK, 256, 0, stream>>>(h_f, proj_f, L[l].g1, L[l].b1,
                                                h_f, h_bf);
    // FFN (WMMA, GELU fused in first GEMM epilogue)
    gemm(h_bf, wT[l][7], L[l].bf1, nullptr, ff_bf, M, F_, D_, 1, 1.f);
    gemm(ff_bf, wT[l][8], L[l].bf2, proj_f, nullptr, M, D_, F_, 0, 1.f);
    res_ln_kernel<<<(int)TOK, 256, 0, stream>>>(h_f, proj_f, L[l].g2, L[l].b2,
                                                h_f, h_bf);
  }

  // ---- pooler + classifier ----
  pool_kernel<<<B_, 256, 0, stream>>>(h_f, wp, bp, pool_f);
  cls1_kernel<<<B_, 256, 0, stream>>>(pool_f, wc1, bc1, c1_f);
  cls2_kernel<<<B_, 32, 0, stream>>>(c1_f, wc2, bc2, (float*)d_out);
}
